// Attention_29987461661416
// MI455X (gfx1250) — compile-verified, matrix operand order checked by reading, not run on hardware
//
#include <hip/hip_runtime.h>
#include <hip/hip_bf16.h>
#include <math.h>

#define BATCH    4
#define SEQ      2048
#define D_MODEL  1024
#define HEADS    16
#define HEADDIM  64
#define OUT_DIM  (HEADS * HEADDIM)

typedef __attribute__((ext_vector_type(8)))  float  v8f;
typedef __attribute__((ext_vector_type(16))) __bf16 bf16x16;
typedef __attribute__((ext_vector_type(8)))  __bf16 bf16x8;

static __device__ __forceinline__ v8f wmma_bf16(bf16x16 a, bf16x16 b, v8f c) {
    return __builtin_amdgcn_wmma_f32_16x16x32_bf16(
        /*neg_a=*/false, a, /*neg_b=*/false, b,
        /*c_mod=*/(short)0, c, /*reuse_a=*/false, /*reuse_b=*/false);
}

// Load a 16x16x32 A/B fragment from a contiguous bf16 row.
// koff = (lane&16) ? 8 : 0 ; elements 0..7 <- p[koff..], 8..15 <- p[koff+16..]
static __device__ __forceinline__ bf16x16 load_frag(const __bf16* p, int koff) {
    union { bf16x16 v; bf16x8 h[2]; } u;
    u.h[0] = *reinterpret_cast<const bf16x8*>(p + koff);
    u.h[1] = *reinterpret_cast<const bf16x8*>(p + koff + 16);
    return u.v;
}

// ---- DPP16 butterfly reductions within each 16-lane half (no DS, no waits) --
// quad_perm(1,0,3,2)=0xB1 -> xor1 ; quad_perm(2,3,0,1)=0x4E -> xor2
// row_half_mirror=0x141   -> xor7 ; row_mirror=0x140        -> xor15
// masks {1,2,7,15} span GF(2)^4 -> full 16-lane reduction, never crosses lane&16.
template <int CTRL>
static __device__ __forceinline__ float dpp_mov(float x) {
    return __int_as_float(__builtin_amdgcn_update_dpp(
        0, __float_as_int(x), CTRL, 0xf, 0xf, true));
}
static __device__ __forceinline__ float rowmax16(float x) {
    x = fmaxf(x, dpp_mov<0xB1>(x));
    x = fmaxf(x, dpp_mov<0x4E>(x));
    x = fmaxf(x, dpp_mov<0x141>(x));
    x = fmaxf(x, dpp_mov<0x140>(x));
    return x;
}
static __device__ __forceinline__ float rowsum16(float x) {
    x += dpp_mov<0xB1>(x);
    x += dpp_mov<0x4E>(x);
    x += dpp_mov<0x141>(x);
    x += dpp_mov<0x140>(x);
    return x;
}

// ---------------------------------------------------------------------------
// Projection GEMM: Y = X(8192x1024, f32) * W(1024x1024, f32) -> bf16
// LDS tiles staged directly in bf16 (one convert per element, 4 waves reuse).
// MODE 0: out[((b*H+h)*S + s)*64 + d]    (row-major per head, for Q/K)
// MODE 1: out[((b*H+h)*64 + d)*S + s]    (transposed, for V)
// ---------------------------------------------------------------------------
template <int MODE>
__global__ __launch_bounds__(128)
void proj_kernel(const float* __restrict__ X, const float* __restrict__ W,
                 __bf16* __restrict__ out)
{
    __shared__ __bf16 Xs[64][36];
    __shared__ __bf16 Ws[32][68];

    const int lane = threadIdx.x & 31;
    const int wave = threadIdx.x >> 5;
    const int m0   = blockIdx.x * 64;
    const int n0   = blockIdx.y * 64;
    const int lrow = lane & 15;
    const int koff = (lane & 16) ? 8 : 0;

    v8f acc[4] = {};

    for (int kb = 0; kb < D_MODEL; kb += 32) {
        for (int i = threadIdx.x; i < 64 * 32; i += 128) {
            int r = i >> 5, c = i & 31;
            Xs[r][c] = (__bf16)X[(size_t)(m0 + r) * D_MODEL + kb + c];
        }
        for (int i = threadIdx.x; i < 32 * 64; i += 128) {
            int r = i >> 6, c = i & 63;
            Ws[r][c] = (__bf16)W[(size_t)(kb + r) * OUT_DIM + n0 + c];
        }
        // prefetch next K-slab into WGP$ (locality 3 -> WGP scope)
        if (kb + 32 < D_MODEL) {
            const int t_ = threadIdx.x;
            __builtin_prefetch(&X[(size_t)(m0 + (t_ >> 1)) * D_MODEL +
                                  kb + 32 + (t_ & 1) * 16], 0, 3);
            __builtin_prefetch(&W[(size_t)(kb + 32 + (t_ >> 2)) * OUT_DIM +
                                  n0 + (t_ & 3) * 16], 0, 3);
        }
        __syncthreads();

        // A fragment: 16 rows of this wave, K = kb..kb+31
        bf16x16 af;
        const int arow = wave * 16 + lrow;
        #pragma unroll
        for (int i = 0; i < 8; ++i) {
            af[i]     = Xs[arow][koff + i];
            af[8 + i] = Xs[arow][koff + 16 + i];
        }
        #pragma unroll
        for (int nc = 0; nc < 4; ++nc) {
            const int bcol = nc * 16 + lrow;
            bf16x16 bfr;
            #pragma unroll
            for (int i = 0; i < 8; ++i) {
                bfr[i]     = Ws[koff + i][bcol];
                bfr[8 + i] = Ws[koff + 16 + i][bcol];
            }
            acc[nc] = wmma_bf16(af, bfr, acc[nc]);
        }
        __syncthreads();
    }

    const int h = blockIdx.y; // 64-col N-tile == head
    #pragma unroll
    for (int nc = 0; nc < 4; ++nc) {
        #pragma unroll
        for (int j = 0; j < 8; ++j) {
            const int mloc = wave * 16 + j + ((lane & 16) ? 8 : 0);
            const int m = m0 + mloc;
            const int b = m / SEQ, s = m % SEQ;
            const int d = nc * 16 + lrow;
            const __bf16 val = (__bf16)acc[nc][j];
            size_t idx;
            if (MODE == 0)
                idx = (((size_t)(b * HEADS + h) * SEQ) + s) * HEADDIM + d;
            else
                idx = (((size_t)(b * HEADS + h) * HEADDIM) + d) * SEQ + s;
            out[idx] = val;
        }
    }
}

// ---------------------------------------------------------------------------
// Flash attention: grid (S/64, H, B), block 128 (4 waves x 16 q-rows).
// qw/kw: [B,H,S,64] bf16 ; vwT: [B,H,64,S] bf16 ; out: [B,S,H*64] f32
// ---------------------------------------------------------------------------
__global__ __launch_bounds__(128)
void attn_kernel(const __bf16* __restrict__ qw, const __bf16* __restrict__ kw,
                 const __bf16* __restrict__ vwT,
                 const float* __restrict__ v_mask, const float* __restrict__ q_mask,
                 float* __restrict__ out)
{
    __shared__ __bf16 Plds[4][16][36]; // per-wave P tile (16 rows x 32 keys)

    const int lane  = threadIdx.x & 31;
    const int wave  = threadIdx.x >> 5;
    const int qbase = blockIdx.x * 64;
    const int h     = blockIdx.y;
    const int b     = blockIdx.z;
    const int lrow  = lane & 15;
    const int koff  = (lane & 16) ? 8 : 0;
    const int rhalf = (lane & 16) ? 8 : 0;
    const int wrow0 = qbase + wave * 16;      // first q-row owned by this wave

    // Q fragments for this wave's 16 rows (d = 0..31 and 32..63)
    const __bf16* qptr =
        qw + (((size_t)(b * HEADS + h) * SEQ) + wrow0 + lrow) * HEADDIM;
    const bf16x16 qa0 = load_frag(qptr, koff);
    const bf16x16 qa1 = load_frag(qptr + 32, koff);

    v8f  o[4] = {};
    float mstat[8], lstat[8];
    #pragma unroll
    for (int j = 0; j < 8; ++j) { mstat[j] = -1e30f; lstat[j] = 0.0f; }

    const size_t kbase = ((size_t)(b * HEADS + h)) * SEQ * HEADDIM;
    const size_t vbase = ((size_t)(b * HEADS + h)) * HEADDIM * SEQ;

    const int qlast = qbase + 63;
    for (int t = 0; t <= qlast; t += 32) {
        // wave-uniform causal test, forced scalar -> s_cbranch not exec-mask
        const bool diag =
            __builtin_amdgcn_readfirstlane((int)(t + 31 > wrow0)) != 0;

        // prefetch next key tile (K rows + V rows) into WGP$ one iter ahead
        const int tn = t + 32;
        if (tn <= qlast) {
            __builtin_prefetch(kw + kbase + (size_t)(tn + lane) * HEADDIM, 0, 3);
            __builtin_prefetch(vwT + vbase + (size_t)lane * SEQ + tn, 0, 3);
            __builtin_prefetch(vwT + vbase + (size_t)(lane + 32) * SEQ + tn, 0, 3);
        }

        // ---- scores for 32 keys (two 16-key chunks), contraction d=64 ----
        v8f sc[2];
        #pragma unroll
        for (int c = 0; c < 2; ++c) {
            const int key = t + c * 16 + lrow;
            const __bf16* kptr = kw + kbase + (size_t)key * HEADDIM;
            const bf16x16 kb0 = load_frag(kptr, koff);
            const bf16x16 kb1 = load_frag(kptr + 32, koff);
            v8f z = {};
            z = wmma_bf16(qa0, kb0, z);
            z = wmma_bf16(qa1, kb1, z);
            const float km   = v_mask[b * SEQ + key];
            const float addm = (1.0f - km) * 1e10f;
            if (diag) {
                #pragma unroll
                for (int j = 0; j < 8; ++j) {
                    const int qr = wrow0 + j + rhalf;
                    float sv = z[j] * 0.125f - addm;   // 1/sqrt(64)
                    if (key > qr) sv -= 1e10f;         // causal
                    z[j] = sv;
                }
            } else {
                #pragma unroll
                for (int j = 0; j < 8; ++j)
                    z[j] = z[j] * 0.125f - addm;
            }
            sc[c] = z;
        }

        // ---- online softmax (DPP16 reductions, per 16-lane half) ----
        float alpha[8];
        #pragma unroll
        for (int j = 0; j < 8; ++j) {
            const float tmax = rowmax16(fmaxf(sc[0][j], sc[1][j]));
            const float mn   = fmaxf(mstat[j], tmax);
            alpha[j] = __expf(mstat[j] - mn);
            mstat[j] = mn;
        }

        #pragma unroll
        for (int j = 0; j < 8; ++j) {
            const float p0 = __expf(sc[0][j] - mstat[j]);
            const float p1 = __expf(sc[1][j] - mstat[j]);
            sc[0][j] = p0;
            sc[1][j] = p1;
            const float rsum = rowsum16(p0 + p1);
            lstat[j] = lstat[j] * alpha[j] + rsum;
        }
        #pragma unroll
        for (int nc = 0; nc < 4; ++nc)
            #pragma unroll
            for (int j = 0; j < 8; ++j) o[nc][j] *= alpha[j];

        // ---- C-layout P -> A-layout via per-wave LDS ----
        #pragma unroll
        for (int c = 0; c < 2; ++c)
            #pragma unroll
            for (int j = 0; j < 8; ++j)
                Plds[wave][j + rhalf][c * 16 + lrow] = (__bf16)sc[c][j];

        bf16x16 pa;
        {
            const __bf16* pp = &Plds[wave][lrow][0];
            #pragma unroll
            for (int i = 0; i < 8; ++i) {
                pa[i]     = pp[koff + i];
                pa[8 + i] = pp[koff + 16 + i];
            }
        }

        // ---- O += P * V  (B operand from transposed V: contiguous keys) ----
        #pragma unroll
        for (int nc = 0; nc < 4; ++nc) {
            const int dcol = nc * 16 + lrow;
            const __bf16* vptr = vwT + vbase + (size_t)dcol * SEQ + t;
            const bf16x16 vb = load_frag(vptr, koff);
            o[nc] = wmma_bf16(pa, vb, o[nc]);
        }
    }

    // ---- epilogue: divide by l, apply q_mask, scatter to [B,S,H*64] ----
    #pragma unroll
    for (int nc = 0; nc < 4; ++nc) {
        #pragma unroll
        for (int j = 0; j < 8; ++j) {
            const int s_ = wrow0 + j + rhalf;
            const int d  = nc * 16 + lrow;
            const float val = (o[nc][j] / lstat[j]) * q_mask[b * SEQ + s_];
            out[((size_t)b * SEQ + s_) * OUT_DIM + h * HEADDIM + d] = val;
        }
    }
}

// ---------------------------------------------------------------------------
extern "C" void kernel_launch(void* const* d_in, const int* in_sizes, int n_in,
                              void* d_out, int out_size, void* d_ws, size_t ws_size,
                              hipStream_t stream) {
    const float* q      = (const float*)d_in[0];
    const float* k      = (const float*)d_in[1];
    const float* v      = (const float*)d_in[2];
    const float* v_mask = (const float*)d_in[3];
    const float* q_mask = (const float*)d_in[4];
    const float* Wq     = (const float*)d_in[5];
    const float* Wk     = (const float*)d_in[6];
    const float* Wv     = (const float*)d_in[7];

    const size_t per_proj = (size_t)BATCH * HEADS * SEQ * HEADDIM; // bf16 elements
    __bf16* qw  = (__bf16*)d_ws;
    __bf16* kw  = qw + per_proj;
    __bf16* vwT = kw + per_proj;   // 3 * 16MB = 48MB of workspace

    dim3 gproj((BATCH * SEQ) / 64, OUT_DIM / 64); // 128 x 16
    proj_kernel<0><<<gproj, 128, 0, stream>>>(q, Wq, qw);
    proj_kernel<0><<<gproj, 128, 0, stream>>>(k, Wk, kw);
    proj_kernel<1><<<gproj, 128, 0, stream>>>(v, Wv, vwT);

    dim3 gattn(SEQ / 64, HEADS, BATCH); // 32 x 16 x 4
    attn_kernel<<<gattn, 128, 0, stream>>>(qw, kw, vwT, v_mask, q_mask,
                                           (float*)d_out);
}
